// Relation_aware_Pos_28724741276300
// MI455X (gfx1250) — matrix operand order repaired and checked by reading, not compiled, not run on hardware
//
#include <hip/hip_runtime.h>

typedef __attribute__((ext_vector_type(16))) _Float16 v16h;
typedef __attribute__((ext_vector_type(8)))  float    v8f;

constexpr int Bn = 8, Ln = 512, Dn = 300, En = 64, Hn = 64;

// Pack 16 consecutive fp32 values into a f16 fragment (sequential element order).
__device__ inline v16h cvt16(float4 r0, float4 r1, float4 r2, float4 r3) {
    v16h a;
    a[0]  = (_Float16)r0.x; a[1]  = (_Float16)r0.y; a[2]  = (_Float16)r0.z; a[3]  = (_Float16)r0.w;
    a[4]  = (_Float16)r1.x; a[5]  = (_Float16)r1.y; a[6]  = (_Float16)r1.z; a[7]  = (_Float16)r1.w;
    a[8]  = (_Float16)r2.x; a[9]  = (_Float16)r2.y; a[10] = (_Float16)r2.z; a[11] = (_Float16)r2.w;
    a[12] = (_Float16)r3.x; a[13] = (_Float16)r3.y; a[14] = (_Float16)r3.z; a[15] = (_Float16)r3.w;
    return a;
}

// ---------------------------------------------------------------------------
// Kernel 1: ew[b,i,j] = relu(edge[b,i,j,:] @ W_map) . w_ew
// Transposed formulation: emT = W_mapT @ edgeT via WMMA f16.
//   A = W_mapT (constant, 4 h-tiles x 2 K-chunks, built once per wave)
//   B = edgeT  (lane n = row j0+n, 16 consecutive floats per chunk-half)
//   C layout: lane -> j, VGPR/half -> h  ==> w_ew dot is per-lane registers,
//   only ONE xor-16 shuffle per 16-j tile (vs 32 serialized bpermutes before).
// One block per (b,i) row; 8 waves x 4 j-tiles each.
// ---------------------------------------------------------------------------
__global__ __launch_bounds__(256) void k_edge_ew(const float* __restrict__ edge,
                                                 const float* __restrict__ Wmap,
                                                 const float* __restrict__ wew,
                                                 float* __restrict__ ewout) {
    __shared__ _Float16 sWT[Hn * En];   // W_mapT: sWT[h*En + e]
    const int t = threadIdx.x;
    for (int idx = t; idx < En * Hn; idx += 256) {
        const int e = idx >> 6, h = idx & 63;      // idx = e*Hn + h
        sWT[h * En + e] = (_Float16)Wmap[idx];
    }
    __syncthreads();

    const int lane = t & 31, w = t >> 5;
    const int half = lane >> 4;               // 0: lanes 0-15, 1: lanes 16-31
    const int n    = lane & 15;
    const int kbA  = half ? 16 : 0;           // A K base; this code's contiguous k order, k = 16*half + s
    const int kbB  = half ? 16 : 0;           // B K base; same contiguous k order as A

    // A fragments: Af[mt][c] holds W_mapT[mt*16 + m][c*32 + k], m = lane row.
    v16h Af[4][2];
#pragma unroll
    for (int mt = 0; mt < 4; ++mt)
#pragma unroll
        for (int c = 0; c < 2; ++c) {
#pragma unroll
            for (int tt = 0; tt < 16; ++tt) {
                const int k = c * 32 + kbA + tt;
                Af[mt][c][tt] = sWT[(mt * 16 + n) * En + k];
            }
        }
    // Per-lane w_ew weights: this lane's h set is {mt*16 + 8*half + r}.
    float wv[4][8];
#pragma unroll
    for (int mt = 0; mt < 4; ++mt)
#pragma unroll
        for (int r = 0; r < 8; ++r) wv[mt][r] = wew[mt * 16 + 8 * half + r];

    const int bi = blockIdx.x;  // b*L + i
    const float* rowbase = edge + (size_t)bi * Ln * En;
    float* ewrow = ewout + (size_t)bi * Ln;

    for (int jt = w; jt < Ln / 16; jt += 8) {
        const int j0 = jt * 16;
        const float* brow = rowbase + (size_t)(j0 + n) * En;
        if (jt + 8 < Ln / 16)  // prefetch this wave's next tile row (HBM stream)
            __builtin_prefetch(brow + (size_t)128 * En, 0, 3);

        v8f acc0 = {}, acc1 = {}, acc2 = {}, acc3 = {};
#pragma unroll
        for (int c = 0; c < 2; ++c) {
            const float4* p = reinterpret_cast<const float4*>(brow + c * 32 + kbB);
            v16h bf = cvt16(p[0], p[1], p[2], p[3]);
            acc0 = __builtin_amdgcn_wmma_f32_16x16x32_f16(false, Af[0][c], false, bf, (short)0, acc0, false, false);
            acc1 = __builtin_amdgcn_wmma_f32_16x16x32_f16(false, Af[1][c], false, bf, (short)0, acc1, false, false);
            acc2 = __builtin_amdgcn_wmma_f32_16x16x32_f16(false, Af[2][c], false, bf, (short)0, acc2, false, false);
            acc3 = __builtin_amdgcn_wmma_f32_16x16x32_f16(false, Af[3][c], false, bf, (short)0, acc3, false, false);
        }
        // relu + w_ew dot over this lane's 32 h values (pure VALU).
        float s = 0.f;
#pragma unroll
        for (int r = 0; r < 8; ++r) {
            s = fmaf(fmaxf(acc0[r], 0.f), wv[0][r], s);
            s = fmaf(fmaxf(acc1[r], 0.f), wv[1][r], s);
            s = fmaf(fmaxf(acc2[r], 0.f), wv[2][r], s);
            s = fmaf(fmaxf(acc3[r], 0.f), wv[3][r], s);
        }
        // Combine the two halves (h 0-7 vs 8-15 of each h-tile): one shuffle.
        s += __shfl_xor(s, 16, 32);
        if (lane < 16) ewrow[j0 + n] = s;
    }
}

// ---------------------------------------------------------------------------
// Kernel 2: h = x @ W_eto + b_eto   (small fp32 GEMM; W_eto resident in L2)
// ---------------------------------------------------------------------------
__global__ __launch_bounds__(320) void k_xw(const float* __restrict__ x,
                                            const float* __restrict__ Weto,
                                            const float* __restrict__ beto,
                                            float* __restrict__ h) {
    __shared__ float sx[Dn];
    const int row = blockIdx.x;  // b*L + i
    const int t   = threadIdx.x;
    if (t < Dn) sx[t] = x[(size_t)row * Dn + t];
    __syncthreads();
    if (t < Dn) {
        float acc = beto[t];
        for (int k = 0; k < Dn; ++k) acc = fmaf(sx[k], Weto[k * Dn + t], acc);
        h[(size_t)row * Dn + t] = acc;
    }
}

// ---------------------------------------------------------------------------
// Kernel 3: per-row abs-mean normalization + adjacency-masked softmax (in place)
// ---------------------------------------------------------------------------
__global__ __launch_bounds__(512) void k_softmax(const int* __restrict__ adj,
                                                 float* __restrict__ att) {
    __shared__ float red[Ln];
    const int row = blockIdx.x;
    const int t   = threadIdx.x;
    float* arow = att + (size_t)row * Ln;
    const int* adjrow = adj + (size_t)row * Ln;

    const float ew = arow[t];
    red[t] = ew; __syncthreads();
    for (int s = Ln / 2; s > 0; s >>= 1) { if (t < s) red[t] += red[t + s]; __syncthreads(); }
    const float mean = red[0] / (float)Ln;
    __syncthreads();

    const float rm = fabsf(mean) + 1e-10f;
    const float score = (adjrow[t] != 0) ? (ew / rm) : -9e15f;

    red[t] = score; __syncthreads();
    for (int s = Ln / 2; s > 0; s >>= 1) { if (t < s) red[t] = fmaxf(red[t], red[t + s]); __syncthreads(); }
    const float mx = red[0];
    __syncthreads();

    const float e = __expf(score - mx);
    red[t] = e; __syncthreads();
    for (int s = Ln / 2; s > 0; s >>= 1) { if (t < s) red[t] += red[t + s]; __syncthreads(); }
    const float denom = red[0];
    __syncthreads();

    arow[t] = e / denom;
}

// ---------------------------------------------------------------------------
// Kernel 4: h_edge = relu(att @ h)   (WMMA f16, K=512, one wave per 16x16 tile)
// ---------------------------------------------------------------------------
__global__ __launch_bounds__(32) void k_aggr(const float* __restrict__ att,
                                             const float* __restrict__ h,
                                             float* __restrict__ out) {
    const int ntile = blockIdx.x;   // 0..18 (19*16 >= 300)
    const int itile = blockIdx.y;   // 0..31
    const int b     = blockIdx.z;
    const int lane  = threadIdx.x;
    const int n     = lane & 15;
    const int m     = lane & 15;
    const int kbA   = (lane < 16) ? 0 : 8;
    const int kbB   = (lane < 16) ? 0 : 16;
    const int d0 = ntile * 16, i0 = itile * 16;
    const int d  = d0 + n;
    const bool dok = (d < Dn);

    const float* arow = att + (size_t)(b * Ln + i0 + m) * Ln;
    v8f acc = {};
    for (int c = 0; c < Ln / 32; ++c) {
        const float* pa = arow + c * 32 + kbA;
        const float4* p0 = reinterpret_cast<const float4*>(pa);
        const float4* p1 = reinterpret_cast<const float4*>(pa + 16);
        // A 16x32 f16 layout: t<8 -> K=kbA+t, t>=8 -> K=kbA+16+(t-8)
        v16h a = cvt16(p0[0], p0[1], p1[0], p1[1]);
        v16h bf;
#pragma unroll
        for (int e = 0; e < 16; ++e) {
            const int k = c * 32 + kbA + (e < 8 ? e : e + 8);
            bf[e] = dok ? (_Float16)h[(size_t)(b * Ln + k) * Dn + d] : (_Float16)0.f;
        }
        acc = __builtin_amdgcn_wmma_f32_16x16x32_f16(false, a, false, bf, (short)0, acc, false, false);
    }
    if (dok) {
#pragma unroll
        for (int r = 0; r < 8; ++r) {
            const int i = i0 + r + ((lane >= 16) ? 8 : 0);
            out[(size_t)(b * Ln + i) * Dn + d] = fmaxf(acc[r], 0.f);
        }
    }
}

// ---------------------------------------------------------------------------
extern "C" void kernel_launch(void* const* d_in, const int* in_sizes, int n_in,
                              void* d_out, int out_size, void* d_ws, size_t ws_size,
                              hipStream_t stream) {
    const float* x    = (const float*)d_in[0];
    const int*   adj  = (const int*)d_in[1];
    const float* edge = (const float*)d_in[2];
    const float* Wmap = (const float*)d_in[3];
    const float* wew  = (const float*)d_in[4];
    const float* Weto = (const float*)d_in[5];
    const float* beto = (const float*)d_in[6];

    float* out    = (float*)d_out;
    float* h_edge = out;                              // [B,L,D]
    float* att    = out + (size_t)Bn * Ln * Dn;       // [B,L,L] (ew staged here, softmaxed in place)
    float* h      = (float*)d_ws;                     // [B,L,D] scratch

    k_edge_ew<<<Bn * Ln, 256, 0, stream>>>(edge, Wmap, wew, att);
    k_xw<<<Bn * Ln, 320, 0, stream>>>(x, Weto, beto, h);
    k_softmax<<<Bn * Ln, Ln, 0, stream>>>(adj, att);
    dim3 g4((Dn + 15) / 16, Ln / 16, Bn);
    k_aggr<<<g4, 32, 0, stream>>>(att, h, h_edge);
}
